// TinyNVFP4Linear_22582938043081
// MI455X (gfx1250) — matrix operand order repaired and checked by reading, not compiled, not run on hardware
//
#include <hip/hip_runtime.h>

typedef __attribute__((ext_vector_type(16))) _Float16 v16h;
typedef __attribute__((ext_vector_type(8)))  float    v8f;
typedef int v4i __attribute__((vector_size(16)));

#define E2M1_MAX 6.0f
#define E4M3_MAX 448.0f
#define TINYF 1e-30f

// ---------------- reference quantizers (RNE via rintf) ----------------

__device__ __forceinline__ float quant_e2m1(float x) {
    float s  = (x > 0.0f) ? 1.0f : ((x < 0.0f) ? -1.0f : 0.0f);
    float ax = fminf(fabsf(x), E2M1_MAX);
    float e  = floorf(__log2f(fmaxf(ax, TINYF)));
    e = fminf(fmaxf(e, 0.0f), 2.0f);
    float step = exp2f(e);
    float q = rintf(ax / step * 2.0f) * 0.5f * step;
    return s * q;
}

__device__ __forceinline__ float quant_e4m3(float x) {
    float s  = (x > 0.0f) ? 1.0f : ((x < 0.0f) ? -1.0f : 0.0f);
    float ax = fminf(fabsf(x), E4M3_MAX);
    float e  = floorf(__log2f(fmaxf(ax, TINYF)));
    e = fminf(fmaxf(e, -6.0f), 8.0f);
    float step = exp2f(e);
    float q = rintf(ax / step * 8.0f) * 0.125f * step;
    return s * q;
}

// ---------------- kernel 1: x f32 -> f16 ----------------

struct H8 { _Float16 v[8]; };

__global__ void f32_to_f16_kernel(const float* __restrict__ in,
                                  _Float16* __restrict__ out, size_t n) {
    size_t i = ((size_t)blockIdx.x * blockDim.x + threadIdx.x) * 8;
    if (i >= n) return;
    float4 a = *(const float4*)(in + i);
    float4 b = *(const float4*)(in + i + 4);
    H8 h;
    h.v[0] = (_Float16)a.x; h.v[1] = (_Float16)a.y;
    h.v[2] = (_Float16)a.z; h.v[3] = (_Float16)a.w;
    h.v[4] = (_Float16)b.x; h.v[5] = (_Float16)b.y;
    h.v[6] = (_Float16)b.z; h.v[7] = (_Float16)b.w;
    *(H8*)(out + i) = h;
}

// ---------------- kernel 2: nvfp4 weight dequant -> f16 ----------------

__global__ void dequant_kernel(const float* __restrict__ w,
                               const float* __restrict__ pbamax,
                               const float* __restrict__ gamax,
                               _Float16* __restrict__ wh,
                               int nblocks) {
    int b = blockIdx.x * blockDim.x + threadIdx.x;
    if (b >= nblocks) return;
    float gscale = gamax[0] * (1.0f / (E2M1_MAX * E4M3_MAX));
    float bs = quant_e4m3(pbamax[b] / E2M1_MAX / gscale) * gscale;
    const float* wp = w + (size_t)b * 16;
    _Float16* op = wh + (size_t)b * 16;
    float den = fmaxf(bs, TINYF);
#pragma unroll
    for (int i = 0; i < 16; i++) {
        float wq = (bs > 0.0f) ? (wp[i] / den) : 0.0f;
        op[i] = (_Float16)(quant_e2m1(wq) * bs);
    }
}

// ---------------- kernel 3: f16 WMMA GEMM: C(TxO) = A(TxI) * B(OxI)^T ---

#define BM 128
#define BN 128
#define BK 64
#define LDT (BK + 8)   // padded LDS row stride (halfs): 72 halfs = 36 dwords

#if defined(__AMDGCN__) && __has_builtin(__builtin_amdgcn_global_load_async_to_lds_b128)
#define USE_ASYNC_LDS 1
#else
#define USE_ASYNC_LDS 0
#endif

#define GLB_AS __attribute__((address_space(1)))
#define LDS_AS __attribute__((address_space(3)))

__device__ __forceinline__ void wait_async0() {
#if defined(__AMDGCN__)
#if __has_builtin(__builtin_amdgcn_s_wait_asynccnt)
    __builtin_amdgcn_s_wait_asynccnt(0);
#else
    asm volatile("s_wait_asynccnt 0x0" ::: "memory");
#endif
#endif
}

union AV { uint4 u[2]; v16h h; };

__global__ __launch_bounds__(256)
void gemm_wmma_f16(const _Float16* __restrict__ A,   // T x I (row-major, K contig)
                   const _Float16* __restrict__ B,   // O x I (row-major, K contig)
                   float* __restrict__ C,            // T x O
                   int T, int O, int I) {
    __shared__ __align__(16) _Float16 As[2][BM * LDT];
    __shared__ __align__(16) _Float16 Bs[2][BN * LDT];

    const int tid  = threadIdx.x;
    const int wave = tid >> 5;
    const int lane = tid & 31;

    const int mTile = blockIdx.y * BM;
    const int nTile = blockIdx.x * BN;

    // 8 waves: 4 along M (32 rows each) x 2 along N (64 cols each)
    const int wM = (wave & 3) * 32;
    const int wN = (wave >> 2) * 64;
    const int half16 = lane >> 4;   // 0/1: upper/lower lane group
    const int l15    = lane & 15;

    v8f acc[2][4];
#pragma unroll
    for (int i = 0; i < 2; i++)
#pragma unroll
        for (int j = 0; j < 4; j++)
            acc[i][j] = (v8f){0.f,0.f,0.f,0.f,0.f,0.f,0.f,0.f};

    // staging layout: 256 threads; each tile row = BK halfs = 128B = 8 x 16B
    const int lr = tid >> 3;        // row within 32-row pass
    const int lc = (tid & 7) * 8;   // half-element column (16B granule)
    const _Float16* Ag = A + (size_t)(mTile + lr) * I + lc;
    const _Float16* Bg = B + (size_t)(nTile + lr) * I + lc;

#if USE_ASYNC_LDS
    // async DMA: global -> LDS, no VGPR round-trip, tracked by ASYNCcnt
    auto stage = [&](int buf, int k0) {
#pragma unroll
        for (int p = 0; p < 4; p++) {
            __builtin_amdgcn_global_load_async_to_lds_b128(
                (GLB_AS v4i*)(Ag + (size_t)(p * 32) * I + k0),
                (LDS_AS v4i*)(&As[buf][(p * 32 + lr) * LDT + lc]), 0, 0);
            __builtin_amdgcn_global_load_async_to_lds_b128(
                (GLB_AS v4i*)(Bg + (size_t)(p * 32) * I + k0),
                (LDS_AS v4i*)(&Bs[buf][(p * 32 + lr) * LDT + lc]), 0, 0);
        }
        if (k0 + BK < I) {  // warm L2 for the tile after this one
            __builtin_prefetch(Ag + k0 + BK, 0, 0);
            __builtin_prefetch(Bg + k0 + BK, 0, 0);
        }
    };
#else
    uint4 ra[4], rb[4];
    auto gload = [&](int k0) {
#pragma unroll
        for (int p = 0; p < 4; p++) {
            ra[p] = *(const uint4*)(Ag + (size_t)(p * 32) * I + k0);
            rb[p] = *(const uint4*)(Bg + (size_t)(p * 32) * I + k0);
        }
    };
    auto commit = [&](int buf) {
#pragma unroll
        for (int p = 0; p < 4; p++) {
            *(uint4*)(&As[buf][(p * 32 + lr) * LDT + lc]) = ra[p];
            *(uint4*)(&Bs[buf][(p * 32 + lr) * LDT + lc]) = rb[p];
        }
    };
#endif

    auto compute = [&](int buf) {
#pragma unroll
        for (int kk = 0; kk < BK; kk += 32) {
            // A fragment (16x32 f16): lane group h holds K in {8h..8h+7, 16+8h..+7}
            v16h afrag[2];
#pragma unroll
            for (int fm = 0; fm < 2; fm++) {
                const _Float16* base = &As[buf][(wM + fm * 16 + l15) * LDT + kk];
                AV t;
                t.u[0] = *(const uint4*)(base + 8 * half16);
                t.u[1] = *(const uint4*)(base + 16 + 8 * half16);
                afrag[fm] = t.h;
            }
            // B fragment (32x16 f16): lane holds col N=l15, K in {8h..8h+7, 16+8h..+7}
            v16h bfrag[4];
#pragma unroll
            for (int fn = 0; fn < 4; fn++) {
                const _Float16* base = &Bs[buf][(wN + fn * 16 + l15) * LDT + kk + 8 * half16];
                AV t;
                t.u[0] = *(const uint4*)(base);
                t.u[1] = *(const uint4*)(base + 16);
                bfrag[fn] = t.h;
            }
#pragma unroll
            for (int fm = 0; fm < 2; fm++)
#pragma unroll
                for (int fn = 0; fn < 4; fn++)
                    acc[fm][fn] = __builtin_amdgcn_wmma_f32_16x16x32_f16(
                        false, afrag[fm], false, bfrag[fn],
                        (short)0, acc[fm][fn], false, false);
        }
    };

#if USE_ASYNC_LDS
    stage(0, 0);
#pragma unroll 1
    for (int k0 = 0; k0 < I; k0 += 2 * BK) {
        // ---- phase A: consume buf 0, DMA next tile into buf 1 ----
        wait_async0();
        __syncthreads();                 // tile k0 fully resident in buf 0
        if (k0 + BK < I) stage(1, k0 + BK);
        compute(0);
        __syncthreads();                 // everyone done reading buf 0
        // ---- phase B: consume buf 1, DMA next tile into buf 0 ----
        wait_async0();
        __syncthreads();
        if (k0 + 2 * BK < I) stage(0, k0 + 2 * BK);
        compute(1);
        __syncthreads();
    }
#else
    gload(0);
#pragma unroll 1
    for (int k0 = 0; k0 < I; k0 += 2 * BK) {
        commit(0);
        __syncthreads();
        if (k0 + BK < I) gload(k0 + BK);
        compute(0);
        __syncthreads();
        commit(1);
        __syncthreads();
        if (k0 + 2 * BK < I) gload(k0 + 2 * BK);
        compute(1);
        __syncthreads();
    }
#endif

    // C layout: VGPR r, lanes 0-15: M=r, N=lane; lanes 16-31: M=8+r, N=lane-16
#pragma unroll
    for (int fm = 0; fm < 2; fm++)
#pragma unroll
        for (int fn = 0; fn < 4; fn++) {
            size_t base = (size_t)(mTile + wM + fm * 16 + half16 * 8) * O
                        + (size_t)(nTile + wN + fn * 16 + l15);
#pragma unroll
            for (int r = 0; r < 8; r++)
                C[base + (size_t)r * O] = acc[fm][fn][r];
        }
}

// ---------------- launch ----------------

extern "C" void kernel_launch(void* const* d_in, const int* in_sizes, int n_in,
                              void* d_out, int out_size, void* d_ws, size_t ws_size,
                              hipStream_t stream) {
    const float* x    = (const float*)d_in[0];
    const float* w    = (const float*)d_in[1];
    const float* pbam = (const float*)d_in[2];
    const float* gam  = (const float*)d_in[3];
    float* out = (float*)d_out;

    const int I = 4096;
    const int O = in_sizes[1] / I;
    const int T = in_sizes[0] / I;

    _Float16* xh = (_Float16*)d_ws;
    _Float16* wh = (_Float16*)((char*)d_ws + (size_t)T * I * sizeof(_Float16));

    size_t nx = (size_t)T * I;
    f32_to_f16_kernel<<<dim3((unsigned)(nx / 8 / 256)), 256, 0, stream>>>(x, xh, nx);

    int nblk = O * (I / 16);
    dequant_kernel<<<dim3((nblk + 255) / 256), 256, 0, stream>>>(w, pbam, gam, wh, nblk);

    dim3 grid(O / BN, T / BM);
    gemm_wmma_f16<<<grid, 256, 0, stream>>>(xh, wh, out, T, O, I);
}
